// NIoULoss_pixel_52905407152953
// MI455X (gfx1250) — matrix operand order repaired and checked by reading, not compiled, not run on hardware
//
#include <hip/hip_runtime.h>

// ---------------------------------------------------------------------------
// NIoU pixel loss for MI455X (gfx1250, wave32, WMMA).
// Memory-bound (~201 MB @ 23.3 TB/s => ~8.6us floor). All bn-contractions run
// on the matrix pipe via V_WMMA_F32_16X16X32_F16; loop body is branch-free
// (EXEC all-ones at WMMA) and VALU-minimized so the SIMDs can keep the
// 23.3 TB/s load stream saturated. Raw v_sqrt_f32 (result goes to f16 anyway).
// ---------------------------------------------------------------------------

typedef _Float16 v16h __attribute__((ext_vector_type(16)));
typedef _Float16 v8h  __attribute__((ext_vector_type(8)));
typedef float    v8f  __attribute__((ext_vector_type(8)));
typedef int      v4i  __attribute__((ext_vector_type(4)));
typedef unsigned v8u  __attribute__((ext_vector_type(8)));

#define EPSF 1e-6f
#define HW_   (512*512)
#define NB_   8             // batch
#define NM_   8             // mask channels
#define ITERS_        8     // 32 pixels per wave iteration
#define PIX_PER_WAVE  (ITERS_*32)
#define PIX_PER_BLOCK (8*PIX_PER_WAVE)

__global__ __launch_bounds__(256) void niou_stage1(
    const float* __restrict__ pd_rgbo, const float* __restrict__ gt_rgbo,
    const int* __restrict__ pd_mask, const int* __restrict__ gt_mask,
    float* __restrict__ ws)
{
  const int tid  = threadIdx.x;
  const int lane = tid & 31;
  const int wv   = tid >> 5;        // wave id within block, 0..7
  const int r    = lane & 15;       // A row / publish row
  const int hi   = lane >> 4;       // 0: K=0..15 half, 1: K=16..31 half
  const int n8   = lane & 7;        // mask channel n
  const bool isGm = (lane & 8) != 0;
  const int b = blockIdx.y;

  // Per-wave staging (wave-synchronous: DS is in-order within a wave).
  // s_ab rows: 0=diff 1=pd_o 2=gt_o 3=ones 4=zeros
  // s_gm rows: 0..7 scratch(pm), 8..15 gm columns, 16 = zeros
  __shared__ __attribute__((aligned(32))) _Float16 s_ab[8][5][32];
  __shared__ __attribute__((aligned(32))) _Float16 s_gm[8][17][32];

  // One-time constant rows (in-order DS => visible to all later loop loads).
  s_ab[wv][3][lane]  = (_Float16)1.0f;
  s_ab[wv][4][lane]  = (_Float16)0.0f;
  s_gm[wv][16][lane] = (_Float16)0.0f;

  const int base = blockIdx.x * PIX_PER_BLOCK + wv * PIX_PER_WAVE;

  // Carried 64-bit pointers; plane strides stay as 24-bit signed immediates.
  const size_t foff = (size_t)b * 4 * HW_ + (size_t)(base + lane);
  const float* pf_pd = pd_rgbo + foff;
  const float* pf_gt = gt_rgbo + foff;
  const int* mroot = isGm ? gt_mask : pd_mask;
  const int* pmsk  = mroot + ((size_t)b * NM_ + n8) * (size_t)HW_
                           + (size_t)(base + hi * 16);

  // Loop-invariant LDS addresses (no per-iter selects => EXEC stays full).
  const int aRow = (r == 0) ? 0 : (r == 2) ? 1 : (r == 3) ? 2 : (r == 1) ? 3 : 4;
  const v8h*  pA0  = (const v8h*)(&s_ab[wv][aRow][0] + hi * 16);
  const v8h*  pA1  = (const v8h*)(&s_ab[wv][aRow][0] + hi * 16 + 8);
  v16h*       pPub = (v16h*)&s_gm[wv][r][0] + hi;                 // publish col
  const v16h* pG   = (const v16h*)&s_gm[wv][isGm ? 16 : (8 + n8)][0] + hi;

  v8f c1 = {};   // rows x [pm | gm]: sim, count, sum(pd_o*pm), gt_sum
  v8f c2 = {};   // rows x [and | 0]: inter

  for (int it = 0; it < ITERS_; ++it) {
    // ---- per-pixel float stage: 1 pixel/lane, coalesced plane loads --------
    {
      float pr = pf_pd[0*HW_], pg = pf_pd[1*HW_], pb = pf_pd[2*HW_], po = pf_pd[3*HW_];
      float gr = pf_gt[0*HW_], gg = pf_gt[1*HW_], gb = pf_gt[2*HW_], go = pf_gt[3*HW_];
      float dr = pr - gr, dg = pg - gg, db = pb - gb;
      // raw v_sqrt_f32: no denormal fix-up sequence; result truncated to f16
      float diff = __builtin_amdgcn_sqrtf(fmaf(dr, dr, fmaf(dg, dg, db * db)));
      s_ab[wv][0][lane] = (_Float16)diff;
      s_ab[wv][1][lane] = (_Float16)__builtin_rintf(po);
      s_ab[wv][2][lane] = (_Float16)__builtin_rintf(go);
    }

    // ---- B1 column: 16 contiguous ints of this lane's mask plane ----------
    // masks are exactly 0/1: (m0 | m1<<16) * 0x3C00 packs two f16 {0,1.0}
    v16h bm;
    {
      const v4i* mp = (const v4i*)pmsk;                 // 4x b128, 64B/lane
      v4i m0 = mp[0], m1 = mp[1], m2 = mp[2], m3 = mp[3];
      int mm[16] = { m0[0],m0[1],m0[2],m0[3], m1[0],m1[1],m1[2],m1[3],
                     m2[0],m2[1],m2[2],m2[3], m3[0],m3[1],m3[2],m3[3] };
      v8u bw;
      #pragma unroll
      for (int k = 0; k < 8; ++k)
        bw[k] = (unsigned)(mm[2*k] | (mm[2*k+1] << 16)) * 0x3C00u;
      bm = __builtin_bit_cast(v16h, bw);
      *pPub = bm;                                        // gm lanes publish
    }

    // ---- gather A tile (contiguous k = 16*hi + s, same order as B) -----------------
    v8h lo = *pA0;
    v8h hh = *pA1;
    v16h a = __builtin_shufflevector(lo, hh,
               0,1,2,3,4,5,6,7,8,9,10,11,12,13,14,15);

    // ---- B2 = AND columns: gm lanes read the zero row => b2==0 for them ----
    v16h g  = *pG;
    v16h b2 = bm * g;            // 0/1 product == logical AND, exact

    // D = A x B + C, f32 accumulate (no divergence anywhere in the loop)
    c1 = __builtin_amdgcn_wmma_f32_16x16x32_f16(false, a, false, bm,
                                                (short)0, c1, false, false);
    c2 = __builtin_amdgcn_wmma_f32_16x16x32_f16(false, a, false, b2,
                                                (short)0, c2, false, false);

    pf_pd += 32; pf_gt += 32; pmsk += 32;
  }

  // C/D tile layout: VGPR m, lanes 0..15 -> (M=m, N=lane).
  float* __restrict__ sim   = ws;
  float* __restrict__ cnt   = ws + 64;
  float* __restrict__ pmo   = ws + 128;   // sum(pd_o * pm)
  float* __restrict__ gts   = ws + 192;
  float* __restrict__ inter = ws + 256;
  if (lane < 8) {
    atomicAdd(&sim[b * 8 + lane],   c1[0]);
    atomicAdd(&cnt[b * 8 + lane],   c1[1]);
    atomicAdd(&pmo[b * 8 + lane],   c1[2]);
    atomicAdd(&inter[b * 8 + lane], c2[2]);
  } else if (lane < 16) {
    atomicAdd(&gts[b * 8 + lane - 8], c1[3]);  // gt_sum sits in cols 8..15
  }
}

__global__ void niou_finalize(const float* __restrict__ ws, float* __restrict__ out)
{
  const int t = threadIdx.x;  // 64 threads = (b,n)
  float simv = ws[t], cntv = ws[64 + t], pmov = ws[128 + t];
  float gtsv = ws[192 + t], intv = ws[256 + t];
  float punish = simv / (cntv + EPSF);
  float xorv = pmov - intv;
  float iou = intv / (gtsv + xorv + EPSF);
  float v = 1.0f - iou + punish;
  __shared__ float red[64];
  red[t] = v;
  __syncthreads();
  #pragma unroll
  for (int s = 32; s > 0; s >>= 1) {
    if (t < s) red[t] += red[t + s];
    __syncthreads();
  }
  if (t == 0) out[0] = red[0] * (1.0f / 64.0f);
}

extern "C" void kernel_launch(void* const* d_in, const int* in_sizes, int n_in,
                              void* d_out, int out_size, void* d_ws, size_t ws_size,
                              hipStream_t stream)
{
  (void)in_sizes; (void)n_in; (void)out_size; (void)ws_size;
  const float* pd_rgbo = (const float*)d_in[0];
  const float* gt_rgbo = (const float*)d_in[1];
  const int*   pd_mask = (const int*)d_in[2];
  const int*   gt_mask = (const int*)d_in[3];
  float* ws  = (float*)d_ws;
  float* out = (float*)d_out;

  hipMemsetAsync(ws, 0, 320 * sizeof(float), stream);
  dim3 grid(HW_ / PIX_PER_BLOCK, NB_, 1);   // 128 x 8 blocks, 8192 waves
  niou_stage1<<<grid, 256, 0, stream>>>(pd_rgbo, gt_rgbo, pd_mask, gt_mask, ws);
  niou_finalize<<<1, 64, 0, stream>>>(ws, out);
}
